// Encoder_6313601925238
// MI455X (gfx1250) — matrix operand order repaired and checked by reading, not compiled, not run on hardware
//
#include <hip/hip_runtime.h>
#include <hip/hip_bf16.h>
#include <math.h>

// ---------------------------------------------------------------------------
// VGAE encoder for MI455X (gfx1250, wave32, WMMA).
// GEMMs use v_wmma_f32_16x16x32_bf16 with weights pre-swizzled into fragment
// order (one v16bf = 2x global_load_b128 per B fragment). Edge aggregation
// uses f32 global atomics resident in the 192MB L2.
// ---------------------------------------------------------------------------

#define N_NODES 100000
#define IN_F    256
#define HID_F   128
#define OUT_F   64
#define NCOL    128            // both GEMMs produce 128 columns
#define NCOLT   (NCOL / 16)    // 8 column tiles
#define KT1     (IN_F / 32)    // 8 K-tiles for GEMM1
#define KT2     (HID_F / 32)   // 4 K-tiles for GEMM2

typedef __attribute__((ext_vector_type(16))) __bf16 v16bf;
typedef __attribute__((ext_vector_type(8)))  float  v8f;

// ---------------- elementwise helpers ----------------

__global__ void k_zero_f32(float* __restrict__ p, int n) {
    int i = blockIdx.x * blockDim.x + threadIdx.x;
    if (i < n) p[i] = 0.0f;
}

__global__ void k_degree(const int* __restrict__ src, float* __restrict__ deg, int E) {
    int i = blockIdx.x * blockDim.x + threadIdx.x;
    if (i < E) atomicAdd(&deg[src[i]], 1.0f);
}

__global__ void k_norm(float* __restrict__ deg_norm, int n) {
    int i = blockIdx.x * blockDim.x + threadIdx.x;
    if (i < n) deg_norm[i] = 1.0f / fmaxf(deg_norm[i], 1.0f);
}

// ---------------- weight pre-swizzle into WMMA B-fragment order -------------
// Fragment layout: frag[(kTile*NCOLT + colTile)*32 + lane] is a v16bf holding
// B[kTile*32 + (e<8 ? 8*(lane>>4) + e : 16 + 8*(lane>>4) + e-8)][colTile*16 + (lane&15)], e = 0..15.

__global__ void k_prep_wsh(const float* __restrict__ Wsh, __bf16* __restrict__ WshB) {
    int i = blockIdx.x * blockDim.x + threadIdx.x;   // over KT1*NCOLT*512
    if (i < KT1 * NCOLT * 512) {
        int e    = i & 15;
        int lane = (i >> 4) & 31;
        int tile = i >> 9;
        int colTile = tile & (NCOLT - 1);
        int kTile   = tile / NCOLT;
        int m = lane & 15, hi = lane >> 4;
        int k = kTile * 32 + (e < 8 ? 8 * hi + e : 16 + 8 * hi + (e - 8));
        int c = colTile * 16 + m;
        WshB[i] = (__bf16)Wsh[k * NCOL + c];
    }
}

// Wcat = [W_mu | W_ls] (128x128), swizzled the same way.
__global__ void k_prep_wcat(const float* __restrict__ Wmu, const float* __restrict__ Wls,
                            __bf16* __restrict__ WcatB) {
    int i = blockIdx.x * blockDim.x + threadIdx.x;   // over KT2*NCOLT*512
    if (i < KT2 * NCOLT * 512) {
        int e    = i & 15;
        int lane = (i >> 4) & 31;
        int tile = i >> 9;
        int colTile = tile & (NCOLT - 1);
        int kTile   = tile / NCOLT;
        int m = lane & 15, hi = lane >> 4;
        int k = kTile * 32 + (e < 8 ? 8 * hi + e : 16 + 8 * hi + (e - 8));
        int c = colTile * 16 + m;
        float v = (c < OUT_F) ? Wmu[k * OUT_F + c] : Wls[k * OUT_F + (c - OUT_F)];
        WcatB[i] = (__bf16)v;
    }
}

// h = relu(agg + b_shared), stored bf16 for GEMM2.
__global__ void k_bias_relu_bf16(const float* __restrict__ G,
                                 const float* __restrict__ b,
                                 __bf16* __restrict__ hbf) {
    int i = blockIdx.x * blockDim.x + threadIdx.x;
    if (i < N_NODES * HID_F) {
        float v = G[i] + b[i & (HID_F - 1)];
        hbf[i] = (__bf16)fmaxf(v, 0.0f);
    }
}

// out = (mu_agg + b_mu) + noise * exp(ls_agg + b_ls)
__global__ void k_final(const float* __restrict__ G,
                        const float* __restrict__ b_mu,
                        const float* __restrict__ b_ls,
                        const float* __restrict__ noise,
                        float* __restrict__ out) {
    int i = blockIdx.x * blockDim.x + threadIdx.x;
    if (i < N_NODES * OUT_F) {
        int n = i >> 6, f = i & (OUT_F - 1);
        float mu = G[(size_t)n * NCOL + f]         + b_mu[f];
        float ls = G[(size_t)n * NCOL + OUT_F + f] + b_ls[f];
        out[i] = mu + noise[i] * __expf(ls);
    }
}

// Per-edge scatter: G[dst] += P[src], 128 floats/edge, one block per edge.
__global__ void k_scatter(const int* __restrict__ src, const int* __restrict__ dst,
                          const float* __restrict__ P, float* __restrict__ G, int E) {
    int e = blockIdx.x;
    int f = threadIdx.x;
    if (e < E) {
        float v = P[(size_t)src[e] * NCOL + f];
        atomicAdd(&G[(size_t)dst[e] * NCOL + f], v);
    }
}

// ---------------- WMMA GEMMs ----------------
// One wave computes one 16x16 C tile; 4 waves / block. C = (A@B) * norm[row].

// GEMM1: A = feat (N x 256 fp32, converted to bf16 in-register), B pre-swizzled.
__global__ void k_gemm1_wmma(const float* __restrict__ A,
                             const __bf16* __restrict__ B,
                             const float* __restrict__ norm,
                             float* __restrict__ P) {
    const int wave = threadIdx.x >> 5;
    const int lane = threadIdx.x & 31;
    const int rowTile = blockIdx.y * 4 + wave;
    if (rowTile * 16 >= N_NODES) return;
    const int row0 = rowTile * 16;
    const int colTile = blockIdx.x;
    const int m  = lane & 15;
    const int hi = lane >> 4;      // lane half selects K sub-range
    const int kb = hi * 8;         // A: K offset within 32-chunk
    v8f acc = {};
    const float* arow  = A + (size_t)(row0 + m) * IN_F;
    const v16bf* Bfrag = (const v16bf*)B + (size_t)colTile * 32 + lane;
#pragma unroll
    for (int kt = 0; kt < KT1; ++kt) {
        const int k = kt * 32;
        v16bf a;
#pragma unroll
        for (int i = 0; i < 8; ++i) {
            a[i]     = (__bf16)arow[k + kb + i];           // K = k+kb+i
            a[i + 8] = (__bf16)arow[k + 16 + kb + i];      // K = k+16+kb+i
        }
        v16bf b = Bfrag[(size_t)kt * NCOLT * 32];          // 2x global_load_b128
        acc = __builtin_amdgcn_wmma_f32_16x16x32_bf16(false, a, false, b,
                                                      (short)0, acc, false, false);
    }
#pragma unroll
    for (int r = 0; r < 8; ++r) {
        int mm = row0 + r + hi * 8;
        P[(size_t)mm * NCOL + (colTile * 16 + m)] = acc[r] * norm[mm];
    }
}

// GEMM2: A = hbf (N x 128 bf16), B = WcatB pre-swizzled.
__global__ void k_gemm2_wmma(const __bf16* __restrict__ A,
                             const __bf16* __restrict__ B,
                             const float* __restrict__ norm,
                             float* __restrict__ P) {
    const int wave = threadIdx.x >> 5;
    const int lane = threadIdx.x & 31;
    const int rowTile = blockIdx.y * 4 + wave;
    if (rowTile * 16 >= N_NODES) return;
    const int row0 = rowTile * 16;
    const int colTile = blockIdx.x;
    const int m  = lane & 15;
    const int hi = lane >> 4;
    const int kb = hi * 8;
    v8f acc = {};
    const __bf16* arow = A + (size_t)(row0 + m) * HID_F;
    const v16bf* Bfrag = (const v16bf*)B + (size_t)colTile * 32 + lane;
#pragma unroll
    for (int kt = 0; kt < KT2; ++kt) {
        const int k = kt * 32;
        v16bf a;
#pragma unroll
        for (int i = 0; i < 8; ++i) {
            a[i]     = arow[k + kb + i];
            a[i + 8] = arow[k + 16 + kb + i];
        }
        v16bf b = Bfrag[(size_t)kt * NCOLT * 32];
        acc = __builtin_amdgcn_wmma_f32_16x16x32_bf16(false, a, false, b,
                                                      (short)0, acc, false, false);
    }
#pragma unroll
    for (int r = 0; r < 8; ++r) {
        int mm = row0 + r + hi * 8;
        P[(size_t)mm * NCOL + (colTile * 16 + m)] = acc[r] * norm[mm];
    }
}

// ---------------- launch ----------------

static inline size_t align256(size_t x) { return (x + 255) & ~(size_t)255; }

extern "C" void kernel_launch(void* const* d_in, const int* in_sizes, int n_in,
                              void* d_out, int out_size, void* d_ws, size_t ws_size,
                              hipStream_t stream) {
    const float* feat     = (const float*)d_in[0];
    const float* W_shared = (const float*)d_in[1];
    const float* b_shared = (const float*)d_in[2];
    const float* W_mu     = (const float*)d_in[3];
    const float* b_mu     = (const float*)d_in[4];
    const float* W_ls     = (const float*)d_in[5];
    const float* b_ls     = (const float*)d_in[6];
    const float* noise    = (const float*)d_in[7];
    const int*   src      = (const int*)d_in[8];
    const int*   dst      = (const int*)d_in[9];
    const int    E        = in_sizes[8];
    float* out = (float*)d_out;

    // workspace carve-up
    char* ws = (char*)d_ws;
    size_t o = 0;
    float* normv = (float*)(ws + o); o = align256(o + (size_t)N_NODES * 4);
    float* P     = (float*)(ws + o); o = align256(o + (size_t)N_NODES * NCOL * 4);
    float* G     = (float*)(ws + o); o = align256(o + (size_t)N_NODES * NCOL * 4);
    __bf16* hbf  = (__bf16*)(ws + o); o = align256(o + (size_t)N_NODES * HID_F * 2);
    __bf16* WshB = (__bf16*)(ws + o); o = align256(o + (size_t)KT1 * NCOLT * 512 * 2);
    __bf16* WcatB= (__bf16*)(ws + o); o = align256(o + (size_t)KT2 * NCOLT * 512 * 2);
    (void)ws_size; (void)n_in; (void)out_size;

    const int T = 256;
    const int nNG = (N_NODES * NCOL + T - 1) / T;

    // 1) degree -> norm
    k_zero_f32<<<(N_NODES + T - 1) / T, T, 0, stream>>>(normv, N_NODES);
    k_degree  <<<(E + T - 1) / T, T, 0, stream>>>(src, normv, E);
    k_norm    <<<(N_NODES + T - 1) / T, T, 0, stream>>>(normv, N_NODES);

    // 2) bf16 weights in WMMA fragment order
    k_prep_wsh <<<(KT1 * NCOLT * 512 + T - 1) / T, T, 0, stream>>>(W_shared, WshB);
    k_prep_wcat<<<(KT2 * NCOLT * 512 + T - 1) / T, T, 0, stream>>>(W_mu, W_ls, WcatB);

    // 3) conv1: P = (feat @ Wsh) * norm ; G = scatter ; h = relu(G + b)
    dim3 g1(NCOLT, (N_NODES / 16 + 3) / 4);
    k_gemm1_wmma<<<g1, 128, 0, stream>>>(feat, WshB, normv, P);
    k_zero_f32  <<<nNG, T, 0, stream>>>(G, N_NODES * NCOL);
    k_scatter   <<<E, NCOL, 0, stream>>>(src, dst, P, G, E);
    k_bias_relu_bf16<<<nNG, T, 0, stream>>>(G, b_shared, hbf);

    // 4) conv2+conv3 fused: P = (h @ [Wmu|Wls]) * norm ; G = scatter
    k_gemm2_wmma<<<g1, 128, 0, stream>>>(hbf, WcatB, normv, P);
    k_zero_f32  <<<nNG, T, 0, stream>>>(G, N_NODES * NCOL);
    k_scatter   <<<E, NCOL, 0, stream>>>(src, dst, P, G, E);

    // 5) reparameterized sample
    k_final<<<(N_NODES * OUT_F + T - 1) / T, T, 0, stream>>>(
        G, b_mu, b_ls, noise, out);
}
